// InteractionBasedAttentionModel_8787503088332
// MI455X (gfx1250) — matrix operand order repaired and checked by reading, not compiled, not run on hardware
//
#include <hip/hip_runtime.h>
#include <hip/hip_bf16.h>

typedef __attribute__((ext_vector_type(16))) _Float16 v16h;
typedef __attribute__((ext_vector_type(8)))  _Float16 v8h;
typedef __attribute__((ext_vector_type(8)))  float    v8f;

#define NEG_SLOPE 0.2f

// ---------------------------------------------------------------------------
// f32 [K,NOUT] -> f16 transposed [NOUT,K] (weights are tiny; done once/call).
// Transposed layout makes each lane's WMMA B-fragment 32 contiguous bytes.
// ---------------------------------------------------------------------------
__global__ void f32_to_f16_t_kernel(const float* __restrict__ a, _Float16* __restrict__ b,
                                    int K, int NOUT) {
    int i = blockIdx.x * blockDim.x + threadIdx.x;
    if (i < K * NOUT) {
        int k = i / NOUT, c = i - k * NOUT;
        b[c * K + k] = (_Float16)a[i];
    }
}

// ---------------------------------------------------------------------------
// Tiled WMMA GEMM: out[n,NOUT] = pre(in[n,K]) @ W (+ postBias)
//   W16t  : f16 weights, transposed [NOUT][K]
//   pre   : identity, or (x + preBias) -> ELU when PRE_ELU
// Block covers ROWS = 16*MT rows; blockDim = 32*(NOUT/16); each wave owns a
// 16-column tile and iterates MT row-tiles, holding its B fragment in
// registers across them. A fragments are two aligned ds_load_b128 each
// (A layout here: slot s -> K=hi*16+s for all 16 slots, same order as B).
// ---------------------------------------------------------------------------
template <int K, int NOUT, int MT, bool PRE_ELU, bool POST>
__global__ __launch_bounds__(32 * (NOUT / 16))
void gemm_wmma_kernel(const float* __restrict__ in,
                      const float* __restrict__ preBias,
                      const _Float16* __restrict__ W16t,
                      const float* __restrict__ postBias,
                      float* __restrict__ out, int n) {
    constexpr int ROWS = 16 * MT;
    extern __shared__ _Float16 lds[];
    _Float16* As = lds;               // [ROWS][K]
    _Float16* Bs = lds + ROWS * K;    // [NOUT][K]  (transposed weights)

    const int tid  = threadIdx.x;
    const int nth  = blockDim.x;
    const int row0 = blockIdx.x * ROWS;
    const bool full = (row0 + ROWS) <= n;   // block-uniform: no row guards needed

    // ---- stage A tile (vectorized, optional bias+ELU pre-op), f32 -> f16 ----
    for (int i = tid * 4; i < ROWS * K; i += nth * 4) {
        int m = i / K, k = i - m * K;          // k is a multiple of 4
        int gr = row0 + m;
        float4 v;
        if (full || gr < n) v = *(const float4*)(in + (size_t)gr * K + k);
        else                v = make_float4(0.f, 0.f, 0.f, 0.f);
        float f[4] = {v.x, v.y, v.z, v.w};
        _Float16* dst = As + m * K + k;
#pragma unroll
        for (int j = 0; j < 4; ++j) {
            float t = f[j];
            if (PRE_ELU) {
                t += preBias[k + j];
                t = (t > 0.f) ? t : (__expf(t) - 1.f);
            }
            dst[j] = (_Float16)t;
        }
    }
    // ---- stage B (straight contiguous copy, 16B chunks) ----
    {
        const uint4* wsrc = (const uint4*)W16t;
        uint4*       wdst = (uint4*)Bs;
        for (int i = tid; i < (K * NOUT) / 8; i += nth) wdst[i] = wsrc[i];
    }
    __syncthreads();

    const int wave = tid >> 5;
    const int lane = tid & 31;
    const int col0 = wave * 16;       // blockDim guarantees col0 < NOUT
    const int hi   = lane >> 4;       // lane half: K sub-range / D row group
    const int mn   = lane & 15;       // A: row ; B/D: column

    v8f acc[MT];
#pragma unroll
    for (int mt = 0; mt < MT; ++mt) acc[mt] = (v8f){};

#pragma unroll
    for (int kk = 0; kk < K; kk += 32) {
        // B fragment: 16 contiguous f16, 32B aligned
        v16h b = *(const v16h*)(Bs + (col0 + mn) * K + kk + hi * 16);
#pragma unroll
        for (int mt = 0; mt < MT; ++mt) {
            const _Float16* ap = As + (mt * 16 + mn) * K + kk + hi * 16;
            v8h alo = *(const v8h*)(ap);         // slots 0..7  : K = hi*16+s
            v8h ahi = *(const v8h*)(ap + 8);     // slots 8..15 : K = hi*16+s
            v16h a = __builtin_shufflevector(alo, ahi, 0, 1, 2, 3, 4, 5, 6, 7,
                                             8, 9, 10, 11, 12, 13, 14, 15);
            acc[mt] = __builtin_amdgcn_wmma_f32_16x16x32_f16(
                false, a, false, b, (short)0, acc[mt], false, false);
        }
    }

    // ---- epilogue: D layout vgpr r -> row r+8*hi, column = lane&15 ----
    const float pb = POST ? postBias[col0 + mn] : 0.f;
    float* outcol = out + col0 + mn;
    if (full) {
        // fast path: no per-element guards, no exec-mask churn
#pragma unroll
        for (int mt = 0; mt < MT; ++mt) {
#pragma unroll
            for (int r = 0; r < 8; ++r) {
                int gr = row0 + mt * 16 + r + 8 * hi;
                float v = POST ? (acc[mt][r] + pb) : acc[mt][r];
                outcol[(size_t)gr * NOUT] = v;
            }
        }
    } else {
#pragma unroll
        for (int mt = 0; mt < MT; ++mt) {
#pragma unroll
            for (int r = 0; r < 8; ++r) {
                int gr = row0 + mt * 16 + r + 8 * hi;
                float v = POST ? (acc[mt][r] + pb) : acc[mt][r];
                if (gr < n) outcol[(size_t)gr * NOUT] = v;
            }
        }
    }
}

// ---------------------------------------------------------------------------
// Attention scores: a_src[n,h] = dot(xl[n,h,:], att_src[h,:]); same for dst.
// One wave32 per node; xl layout [n,128], head h at cols h*64..h*64+63.
// ---------------------------------------------------------------------------
__global__ void att_scores_kernel(const float* __restrict__ xl,
                                  const float* __restrict__ att_src,
                                  const float* __restrict__ att_dst,
                                  float* __restrict__ as_out,
                                  float* __restrict__ ad_out, int n) {
    int node = blockIdx.x * (blockDim.x >> 5) + (threadIdx.x >> 5);
    int lane = threadIdx.x & 31;
    if (node >= n) return;
    const float* row = xl + (size_t)node * 128;
    float s0 = row[lane] * att_src[lane] + row[lane + 32] * att_src[lane + 32];
    float s1 = row[64 + lane] * att_src[64 + lane] + row[96 + lane] * att_src[96 + lane];
    float d0 = row[lane] * att_dst[lane] + row[lane + 32] * att_dst[lane + 32];
    float d1 = row[64 + lane] * att_dst[64 + lane] + row[96 + lane] * att_dst[96 + lane];
#pragma unroll
    for (int off = 16; off; off >>= 1) {
        s0 += __shfl_xor(s0, off, 32);
        s1 += __shfl_xor(s1, off, 32);
        d0 += __shfl_xor(d0, off, 32);
        d1 += __shfl_xor(d1, off, 32);
    }
    if (lane == 0) {
        as_out[node * 2 + 0] = s0;
        as_out[node * 2 + 1] = s1;
        ad_out[node * 2 + 0] = d0;
        ad_out[node * 2 + 1] = d1;
    }
}

// order-preserving float<->uint key (total order incl. negatives)
__device__ __forceinline__ unsigned f2key(float f) {
    unsigned u = __float_as_uint(f);
    return (u & 0x80000000u) ? ~u : (u | 0x80000000u);
}
__device__ __forceinline__ float key2f(unsigned k) {
    return (k & 0x80000000u) ? __uint_as_float(k & 0x7FFFFFFFu) : __uint_as_float(~k);
}

// ---------------------------------------------------------------------------
// Edge pass 1: alpha = leaky_relu(a_src[src]+a_dst[dst]); segment-max via
// atomicMax on monotone uint key. Edges e>=E are the implicit self-loops.
// ---------------------------------------------------------------------------
__global__ void edge_alpha_max_kernel(const int* __restrict__ ei,
                                      const float* __restrict__ as_,
                                      const float* __restrict__ ad_,
                                      float* __restrict__ alpha,
                                      unsigned* __restrict__ amaxk,
                                      int E, int Nn) {
    int e = blockIdx.x * blockDim.x + threadIdx.x;
    if (e >= E + Nn) return;
    int src, dst;
    if (e < E) { src = ei[e]; dst = ei[E + e]; } else { src = dst = e - E; }
#pragma unroll
    for (int h = 0; h < 2; ++h) {
        float a = as_[src * 2 + h] + ad_[dst * 2 + h];
        a = (a > 0.0f) ? a : NEG_SLOPE * a;
        alpha[(size_t)e * 2 + h] = a;
        atomicMax(&amaxk[dst * 2 + h], f2key(a));
    }
}

// ---------------------------------------------------------------------------
// Edge pass 2: alpha <- exp(alpha - amax[dst]); denom[dst] += alpha (atomic).
// ---------------------------------------------------------------------------
__global__ void edge_exp_sum_kernel(const int* __restrict__ ei,
                                    const unsigned* __restrict__ amaxk,
                                    float* __restrict__ alpha,
                                    float* __restrict__ denom,
                                    int E, int Nn) {
    int e = blockIdx.x * blockDim.x + threadIdx.x;
    if (e >= E + Nn) return;
    int dst = (e < E) ? ei[E + e] : (e - E);
#pragma unroll
    for (int h = 0; h < 2; ++h) {
        float mx = key2f(amaxk[dst * 2 + h]);
        float ex = __expf(alpha[(size_t)e * 2 + h] - mx);
        alpha[(size_t)e * 2 + h] = ex;
        atomicAdd(&denom[dst * 2 + h], ex);
    }
}

// ---------------------------------------------------------------------------
// Edge pass 3: agg[dst,c] += xl[src,c] * alpha[e,h]/(denom[dst,h]+1e-16)
// 128 threads/edge; gathers are L2-resident (xl = 25.6MB << 192MB L2).
// ---------------------------------------------------------------------------
__global__ void edge_aggregate_kernel(const int* __restrict__ ei,
                                      const float* __restrict__ xl,
                                      const float* __restrict__ alpha,
                                      const float* __restrict__ denom,
                                      float* __restrict__ agg,
                                      int E, int Nn) {
    long long t = (long long)blockIdx.x * blockDim.x + threadIdx.x;
    int e = (int)(t >> 7);
    int c = (int)(t & 127);
    if (e >= E + Nn) return;
    int src, dst;
    if (e < E) { src = ei[e]; dst = ei[E + e]; } else { src = dst = e - E; }
    int h = c >> 6;
    float w = alpha[(size_t)e * 2 + h] / (denom[dst * 2 + h] + 1e-16f);
    atomicAdd(&agg[(size_t)dst * 128 + c], xl[(size_t)src * 128 + c] * w);
}

// ---------------------------------------------------------------------------
// Final head: logits = h @ out_w + out_b; Gumbel; softmax over 32 lanes.
// One wave32 per node, lane = output class.
// ---------------------------------------------------------------------------
__global__ void final_head_kernel(const float* __restrict__ hfeat,
                                  const float* __restrict__ out_w,
                                  const float* __restrict__ out_b,
                                  const float* __restrict__ gu,
                                  float* __restrict__ y, int n) {
    int node = blockIdx.x * (blockDim.x >> 5) + (threadIdx.x >> 5);
    int lane = threadIdx.x & 31;
    if (node >= n) return;
    const float* hr = hfeat + (size_t)node * 64;
    float acc = out_b[lane];
#pragma unroll 8
    for (int k = 0; k < 64; ++k) acc += hr[k] * out_w[k * 32 + lane];
    float g = -__logf(-__logf(gu[(size_t)node * 32 + lane] + 1e-20f) + 1e-20f);
    float z = acc + g;  // TEMP = 1
    float mx = z;
#pragma unroll
    for (int off = 16; off; off >>= 1) mx = fmaxf(mx, __shfl_xor(mx, off, 32));
    float ex = __expf(z - mx);
    float sum = ex;
#pragma unroll
    for (int off = 16; off; off >>= 1) sum += __shfl_xor(sum, off, 32);
    y[(size_t)node * 32 + lane] = ex / sum;
}

// ---------------------------------------------------------------------------
// Host launcher
// ---------------------------------------------------------------------------
extern "C" void kernel_launch(void* const* d_in, const int* in_sizes, int n_in,
                              void* d_out, int out_size, void* d_ws, size_t ws_size,
                              hipStream_t stream) {
    const float* x        = (const float*)d_in[0];
    const int*   ei       = (const int*)  d_in[1];
    const float* W0       = (const float*)d_in[2];
    const float* att_src0 = (const float*)d_in[3];
    const float* att_dst0 = (const float*)d_in[4];
    const float* bias0    = (const float*)d_in[5];
    const float* fcw0     = (const float*)d_in[6];
    const float* fcb0     = (const float*)d_in[7];
    const float* W1       = (const float*)d_in[8];
    const float* att_src1 = (const float*)d_in[9];
    const float* att_dst1 = (const float*)d_in[10];
    const float* bias1    = (const float*)d_in[11];
    const float* fcw1     = (const float*)d_in[12];
    const float* fcb1     = (const float*)d_in[13];
    const float* out_w    = (const float*)d_in[14];
    const float* out_b    = (const float*)d_in[15];
    const float* gumbel_u = (const float*)d_in[16];

    const int N  = in_sizes[0] / 64;   // 50000
    const int E  = in_sizes[1] / 2;    // 800000
    const int Et = E + N;              // with self-loops

    // ---- workspace layout ----
    char*  base = (char*)d_ws;
    size_t off  = 0;
    auto alloc = [&](size_t bytes) -> void* {
        void* p = base + off;
        off = (off + bytes + 255) & ~(size_t)255;
        return p;
    };
    // zero-per-layer region (contiguous): amaxk | denom | agg
    unsigned* amaxk = (unsigned*)alloc((size_t)N * 2 * 4);
    float*    denom = (float*)   alloc((size_t)N * 2 * 4);
    float*    agg   = (float*)   alloc((size_t)N * 128 * 4);
    size_t zero_bytes = (size_t)((char*)agg + (size_t)N * 128 * 4 - (char*)amaxk);

    float*    xl    = (float*)   alloc((size_t)N * 128 * 4);
    float*    hbuf  = (float*)   alloc((size_t)N * 64 * 4);
    float*    as_   = (float*)   alloc((size_t)N * 2 * 4);
    float*    ad_   = (float*)   alloc((size_t)N * 2 * 4);
    float*    alpha = (float*)   alloc((size_t)Et * 2 * 4);
    _Float16* W0t   = (_Float16*)alloc(64 * 128 * 2);   // [128][64]
    _Float16* W1t   = (_Float16*)alloc(64 * 128 * 2);   // [128][64]
    _Float16* fcw0t = (_Float16*)alloc(128 * 64 * 2);   // [64][128]
    _Float16* fcw1t = (_Float16*)alloc(128 * 64 * 2);   // [64][128]
    (void)ws_size; (void)n_in; (void)out_size;

    // ---- convert + transpose weights to f16 once per call ----
    {
        dim3 b(256), g((64 * 128 + 255) / 256);
        f32_to_f16_t_kernel<<<g, b, 0, stream>>>(W0,   W0t,   64, 128);
        f32_to_f16_t_kernel<<<g, b, 0, stream>>>(W1,   W1t,   64, 128);
        f32_to_f16_t_kernel<<<g, b, 0, stream>>>(fcw0, fcw0t, 128, 64);
        f32_to_f16_t_kernel<<<g, b, 0, stream>>>(fcw1, fcw1t, 128, 64);
    }

    constexpr int MT = 4;                 // 64 rows per block
    const unsigned gemm_grid = (unsigned)((N + 16 * MT - 1) / (16 * MT));
    const size_t   shmem1 = (size_t)(16 * MT * 64 + 64 * 128) * sizeof(_Float16);   // 24 KB
    const size_t   shmem2 = (size_t)(16 * MT * 128 + 128 * 64) * sizeof(_Float16);  // 32 KB
    const unsigned node_wave_grid = (unsigned)((N + 7) / 8);
    const unsigned edge_grid = (unsigned)((Et + 255) / 256);
    const unsigned aggr_grid = (unsigned)(((long long)Et * 128 + 255) / 256);

    for (int layer = 0; layer < 2; ++layer) {
        const float*    feat = (layer == 0) ? x : hbuf;
        const _Float16* Wt   = (layer == 0) ? W0t : W1t;
        const float*    asw  = (layer == 0) ? att_src0 : att_src1;
        const float*    adw  = (layer == 0) ? att_dst0 : att_dst1;
        const float*    gb   = (layer == 0) ? bias0 : bias1;
        const _Float16* fwt  = (layer == 0) ? fcw0t : fcw1t;
        const float*    fb   = (layer == 0) ? fcb0 : fcb1;

        // zero atomic accumulators (amaxk key 0 == -inf under the monotone map)
        hipMemsetAsync(amaxk, 0, zero_bytes, stream);

        // xl = feat @ W -> [N,128]
        gemm_wmma_kernel<64, 128, MT, false, false><<<gemm_grid, 256, shmem1, stream>>>(
            feat, nullptr, Wt, nullptr, xl, N);

        // per-node attention scores
        att_scores_kernel<<<node_wave_grid, 256, 0, stream>>>(xl, asw, adw, as_, ad_, N);

        // edge softmax (3 passes) + aggregation
        edge_alpha_max_kernel<<<edge_grid, 256, 0, stream>>>(ei, as_, ad_, alpha, amaxk, E, N);
        edge_exp_sum_kernel  <<<edge_grid, 256, 0, stream>>>(ei, amaxk, alpha, denom, E, N);
        edge_aggregate_kernel<<<aggr_grid, 256, 0, stream>>>(ei, xl, alpha, denom, agg, E, N);

        // h = ELU(agg + gat_bias) @ fcw + fcb -> [N,64]
        gemm_wmma_kernel<128, 64, MT, true, true><<<gemm_grid, 128, shmem2, stream>>>(
            agg, gb, fwt, fb, hbuf, N);
    }

    // logits -> gumbel softmax
    final_head_kernel<<<node_wave_grid, 256, 0, stream>>>(
        hbuf, out_w, out_b, gumbel_u, (float*)d_out, N);
}
